// OVDBaseClassifier_5798205849605
// MI455X (gfx1250) — compile-verified, matrix operand order checked by reading, not run on hardware
//
#include <hip/hip_runtime.h>
#include <math.h>

typedef __attribute__((ext_vector_type(16))) _Float16 v16h;
typedef __attribute__((ext_vector_type(8)))  _Float16 v8h;
typedef __attribute__((ext_vector_type(8)))  float    v8f;

#define B_SZ     8
#define KP       1849      // 43*43 patches
#define D        768
#define N_CLS    20
#define N_PAD    32
#define GRID_P   43
#define OUT_W    602
#define EPS_C    1e-8f
#define WAVES    8
#define TILES_M  116       // ceil(1849/16)
#define BLK_PER_B 15       // ceil(116/8)

// ---------------------------------------------------------------------------
// Kernel 1: fused  dot(b,n,k) / (|feats|*|emb| + eps)  via WMMA f16->f32
// grid.x = B_SZ * BLK_PER_B, block = 256 (8 waves), wave32
// ---------------------------------------------------------------------------
__global__ __launch_bounds__(256) void cosim_wmma_kernel(
    const float* __restrict__ feats,   // (8, 1849, 768) f32
    const float* __restrict__ emb,     // (20, 768) f32
    float* __restrict__ cosim)         // (8, 20, 1849) f32  (workspace)
{
    __shared__ _Float16 sEmb[N_PAD][D];      // 48 KB, B-matrix staged as f16
    __shared__ float    sEmbNorm[N_PAD];
    __shared__ float    sFeatNorm[WAVES][16];

    const int tid = threadIdx.x;

    // Stage embeddings f32 -> f16 into LDS, zero-pad rows 20..31
    for (int i = tid; i < N_PAD * D; i += 256) {
        const int n = i / D, k = i - n * D;
        const float v = (n < N_CLS) ? emb[n * D + k] : 0.0f;
        sEmb[n][k] = (_Float16)v;
    }
    // Prototype norms in f32 (exact)
    if (tid < N_PAD) {
        float s = 0.0f;
        if (tid < N_CLS) {
            const float* rp = emb + tid * D;
            for (int k = 0; k < D; ++k) { const float v = rp[k]; s += v * v; }
        }
        sEmbNorm[tid] = sqrtf(s);
    }
    __syncthreads();

    const int wave = tid >> 5;
    const int lane = tid & 31;
    const int b    = blockIdx.x / BLK_PER_B;
    const int tg   = blockIdx.x - b * BLK_PER_B;
    const int tile = tg * WAVES + wave;        // wave-uniform
    const bool active = (tile < TILES_M);
    const int m0 = tile * 16;

    v8f  acc0 = {};      // N = 0..15
    v8f  acc1 = {};      // N = 16..31
    float nrm = 0.0f;    // per-lane partial sum of squares (rows of A)

    if (active) {
        const int rowLocal = lane & 15;
        int row = m0 + rowLocal;
        if (row > KP - 1) row = KP - 1;        // clamp; result discarded later
        const float* rowp = feats + ((size_t)b * KP + row) * D;
        const int hi = (lane >> 4) & 1;        // half-wave select
        const _Float16* bp0 = &sEmb[rowLocal][0];
        const _Float16* bp1 = &sEmb[16 + rowLocal][0];

        for (int k0 = 0; k0 < D; k0 += 32) {
            const int klo = k0 + (hi << 3);    // ISA 16-bit A layout: lo lanes
                                               // K={0..7,16..23}, hi {8..15,24..31}
            const float4 f0 = *(const float4*)(rowp + klo);
            const float4 f1 = *(const float4*)(rowp + klo + 4);
            const float4 f2 = *(const float4*)(rowp + klo + 16);
            const float4 f3 = *(const float4*)(rowp + klo + 20);

            nrm += f0.x*f0.x + f0.y*f0.y + f0.z*f0.z + f0.w*f0.w
                 + f1.x*f1.x + f1.y*f1.y + f1.z*f1.z + f1.w*f1.w
                 + f2.x*f2.x + f2.y*f2.y + f2.z*f2.z + f2.w*f2.w
                 + f3.x*f3.x + f3.y*f3.y + f3.z*f3.z + f3.w*f3.w;

            v16h a;
            a[0]=(_Float16)f0.x;  a[1]=(_Float16)f0.y;  a[2]=(_Float16)f0.z;  a[3]=(_Float16)f0.w;
            a[4]=(_Float16)f1.x;  a[5]=(_Float16)f1.y;  a[6]=(_Float16)f1.z;  a[7]=(_Float16)f1.w;
            a[8]=(_Float16)f2.x;  a[9]=(_Float16)f2.y;  a[10]=(_Float16)f2.z; a[11]=(_Float16)f2.w;
            a[12]=(_Float16)f3.x; a[13]=(_Float16)f3.y; a[14]=(_Float16)f3.z; a[15]=(_Float16)f3.w;

            v16h bb0, bb1;
            ((v8h*)&bb0)[0] = *(const v8h*)(bp0 + klo);
            ((v8h*)&bb0)[1] = *(const v8h*)(bp0 + klo + 16);
            ((v8h*)&bb1)[0] = *(const v8h*)(bp1 + klo);
            ((v8h*)&bb1)[1] = *(const v8h*)(bp1 + klo + 16);

            acc0 = __builtin_amdgcn_wmma_f32_16x16x32_f16(
                false, a, false, bb0, (short)0, acc0, false, false);
            acc1 = __builtin_amdgcn_wmma_f32_16x16x32_f16(
                false, a, false, bb1, (short)0, acc1, false, false);

            if (k0 + 32 < D) __builtin_prefetch(rowp + klo + 32, 0, 0);
        }

        // Row L's squares live in lanes L and L+16 -> combine across half-waves
        const float tot = nrm + __shfl_xor(nrm, 16, 32);
        if (lane < 16) sFeatNorm[wave][lane] = sqrtf(tot);
    }
    __syncthreads();

    if (active) {
        const int nl    = lane & 15;
        const int mside = (lane >> 4) << 3;           // C/D layout: hi lanes M+8
        const float en0 = sEmbNorm[nl];
        const float en1 = sEmbNorm[16 + nl];
        #pragma unroll
        for (int r = 0; r < 8; ++r) {
            const int mloc = r + mside;
            const int m = m0 + mloc;
            if (m < KP) {
                const float fn = sFeatNorm[wave][mloc];
                cosim[((size_t)b * N_CLS + nl) * KP + m] =
                    acc0[r] / (fn * en0 + EPS_C);     // n = nl (< 20 always)
                const int n1 = 16 + nl;
                if (n1 < N_CLS)
                    cosim[((size_t)b * N_CLS + n1) * KP + m] =
                        acc1[r] / (fn * en1 + EPS_C);
            }
        }
    }
}

// ---------------------------------------------------------------------------
// Kernel 2: bicubic (Keys a=-0.5) 43x43 -> 602x602, half-pixel centers,
// per-axis edge renormalization (jax.image.resize semantics). BW-bound store.
// grid = (19*76, 160), block = 256 (32x8 output tile)
// ---------------------------------------------------------------------------
__device__ __forceinline__ float kcub_in(float s)  { return ((1.5f*s - 2.5f)*s)*s + 1.0f; }
__device__ __forceinline__ float kcub_out(float s) { return ((-0.5f*s + 2.5f)*s - 4.0f)*s + 2.0f; }

__global__ __launch_bounds__(256) void bicubic_resize_kernel(
    const float* __restrict__ cosim,   // (160, 43, 43)
    float* __restrict__ out)           // (160, 602, 602)
{
    __shared__ float tile[KP];
    const int img = blockIdx.y;
    const float* src = cosim + (size_t)img * KP;
    for (int i = threadIdx.x; i < KP; i += 256) tile[i] = src[i];
    __syncthreads();

    const int t  = blockIdx.x;
    const int x  = (t % 19) * 32 + (threadIdx.x & 31);
    const int y  = (t / 19) * 8  + (threadIdx.x >> 5);
    if (x >= OUT_W || y >= OUT_W) return;

    const float scale = 1.0f / 14.0f;             // 43/602 exactly

    float px = (x + 0.5f) * scale - 0.5f;
    float fx = floorf(px);
    int   ix = (int)fx - 1;
    float txf = px - fx;
    float wx[4] = { kcub_out(1.0f + txf), kcub_in(txf),
                    kcub_in(1.0f - txf),  kcub_out(2.0f - txf) };

    float py = (y + 0.5f) * scale - 0.5f;
    float fy = floorf(py);
    int   iy = (int)fy - 1;
    float tyf = py - fy;
    float wy[4] = { kcub_out(1.0f + tyf), kcub_in(tyf),
                    kcub_in(1.0f - tyf),  kcub_out(2.0f - tyf) };

    float wxsum = 0.0f;
    #pragma unroll
    for (int i = 0; i < 4; ++i) {
        const int xi = ix + i;
        if (xi >= 0 && xi < GRID_P) wxsum += wx[i];
    }

    float acc = 0.0f, wysum = 0.0f;
    #pragma unroll
    for (int j = 0; j < 4; ++j) {
        const int yi = iy + j;
        if (yi < 0 || yi >= GRID_P) continue;
        float r = 0.0f;
        #pragma unroll
        for (int i = 0; i < 4; ++i) {
            const int xi = ix + i;
            if (xi >= 0 && xi < GRID_P) r += wx[i] * tile[yi * GRID_P + xi];
        }
        acc += wy[j] * r;
        wysum += wy[j];
    }

    out[((size_t)img * OUT_W + y) * OUT_W + x] = acc / (wxsum * wysum);
}

// ---------------------------------------------------------------------------
extern "C" void kernel_launch(void* const* d_in, const int* in_sizes, int n_in,
                              void* d_out, int out_size, void* d_ws, size_t ws_size,
                              hipStream_t stream) {
    const float* feats = (const float*)d_in[0];   // 8*1849*768 f32
    const float* emb   = (const float*)d_in[1];   // 20*768 f32
    float* cosim = (float*)d_ws;                  // 8*20*1849 f32 scratch
    float* out   = (float*)d_out;                 // 8*20*602*602 f32

    cosim_wmma_kernel<<<B_SZ * BLK_PER_B, 256, 0, stream>>>(feats, emb, cosim);

    dim3 rgrid(19 * 76, B_SZ * N_CLS);            // 32x8 tiles, 160 images
    bicubic_resize_kernel<<<rgrid, 256, 0, stream>>>(cosim, out);
}
